// GroupFC_49074296324665
// MI455X (gfx1250) — compile-verified, matrix operand order checked by reading, not run on hardware
//
#include <hip/hip_runtime.h>

typedef __attribute__((ext_vector_type(16))) __bf16          v16bf;
typedef __attribute__((ext_vector_type(8)))  float           v8f;
typedef __attribute__((ext_vector_type(8)))  unsigned short  ush8;
typedef __attribute__((ext_vector_type(4)))  unsigned short  ush4;

#define K_DIM 1024
#define N_DIM 1024
#define BM 128
#define BN 128
#define BK 32

__device__ __forceinline__ unsigned short bf16_rne(float f) {
    unsigned int u = __builtin_bit_cast(unsigned int, f);
    u += 0x7FFFu + ((u >> 16) & 1u);
    return (unsigned short)(u >> 16);
}
__device__ __forceinline__ float bf16_to_f(unsigned short h) {
    unsigned int u = ((unsigned int)h) << 16;
    return __builtin_bit_cast(float, u);
}

// LDS byte offset of a __shared__ object (generic -> AS3 -> 32-bit offset)
__device__ __forceinline__ unsigned lds_off(const void* p) {
    return (unsigned)(unsigned long long)
        (__attribute__((address_space(3))) const void*)p;
}

// CDNA5 async DMA: 16B global -> LDS, tracked by ASYNCcnt (no VGPR data path)
__device__ __forceinline__ void async_copy16(const void* gsrc, unsigned lds_dst) {
    unsigned long long ga = (unsigned long long)gsrc;
    asm volatile("global_load_async_to_lds_b128 %0, %1, off"
                 :: "v"(lds_dst), "v"(ga) : "memory");
}
__device__ __forceinline__ void wait_async0() {
#if __has_builtin(__builtin_amdgcn_s_wait_asynccnt)
    __builtin_amdgcn_s_wait_asynccnt(0);
#else
    asm volatile("s_wait_asynccnt 0x0" ::: "memory");
#endif
}

union FragU { v16bf bf; ush8 u[2]; };

// ---- staging helpers (128x32 fp32 tile <-> hi/lo bf16 LDS planes) ----
__device__ __forceinline__ void load_tile_f32(const float* __restrict__ src,
                                              int r0, int k0, int tid, float4 a[4]) {
    #pragma unroll
    for (int it = 0; it < 4; ++it) {
        int q   = tid + it * 256;          // float4 index 0..1023
        int row = q >> 3;
        int c4  = (q & 7) * 4;
        a[it] = *(const float4*)(src + (size_t)(r0 + row) * K_DIM + k0 + c4);
    }
}
__device__ __forceinline__ void store_split(const float4 a[4],
                                            unsigned short* __restrict__ sh,
                                            unsigned short* __restrict__ sl, int tid) {
    #pragma unroll
    for (int it = 0; it < 4; ++it) {
        int q   = tid + it * 256;
        int row = q >> 3;
        int c4  = (q & 7) * 4;
        float f[4] = {a[it].x, a[it].y, a[it].z, a[it].w};
        ush4 hi, lo;
        #pragma unroll
        for (int j = 0; j < 4; ++j) {
            unsigned short h = bf16_rne(f[j]);
            hi[j] = h;
            lo[j] = bf16_rne(f[j] - bf16_to_f(h));
        }
        *(ush4*)&sh[row * BK + c4] = hi;
        *(ush4*)&sl[row * BK + c4] = lo;
    }
}
__device__ __forceinline__ void issue_W_async(const unsigned short* __restrict__ Whi,
                                              const unsigned short* __restrict__ Wlo,
                                              const unsigned short* sWh,
                                              const unsigned short* sWl,
                                              int n0, int k0, int tid) {
    #pragma unroll
    for (int it = 0; it < 2; ++it) {
        int c    = tid + it * 256;          // 16B-chunk id, 0..511
        int row  = c >> 2;
        int koff = (c & 3) * 8;             // ushort offset within row
        size_t g = (size_t)(n0 + row) * K_DIM + k0 + koff;
        unsigned l = (unsigned)(row * BK + koff) * 2u;
        async_copy16(Whi + g, lds_off(sWh) + l);
        async_copy16(Wlo + g, lds_off(sWl) + l);
    }
}

// ---- prep pass: W fp32 -> hi/lo bf16 planes in workspace (4 MB total) ----
__global__ __launch_bounds__(256) void wconvert(
        const float* __restrict__ W, unsigned short* __restrict__ Whi,
        unsigned short* __restrict__ Wlo) {
    const int q = blockIdx.x * 256 + threadIdx.x;      // float4 index
    const float4 v = *(const float4*)(W + (size_t)q * 4);
    float f[4] = {v.x, v.y, v.z, v.w};
    ush4 hi, lo;
    #pragma unroll
    for (int j = 0; j < 4; ++j) {
        unsigned short h = bf16_rne(f[j]);
        hi[j] = h;
        lo[j] = bf16_rne(f[j] - bf16_to_f(h));
    }
    *(ush4*)(Whi + (size_t)q * 4) = hi;
    *(ush4*)(Wlo + (size_t)q * 4) = lo;
}

template <bool PRECONV>
__global__ __launch_bounds__(256) void groupfc_gemm(
        const float* __restrict__ A,            // [M, K] fp32
        const float* __restrict__ Wf,           // [N, K] fp32 (fallback path)
        const unsigned short* __restrict__ Whi, // [N, K] bf16 hi plane
        const unsigned short* __restrict__ Wlo, // [N, K] bf16 lo plane
        const float* __restrict__ bias,         // [N]
        float* __restrict__ C) {                // [M, N]
    // ping-pong LDS buffers: 2 x 32 KB
    __shared__ __align__(16) unsigned short sAh[2][BM * BK];
    __shared__ __align__(16) unsigned short sAl[2][BM * BK];
    __shared__ __align__(16) unsigned short sWh[2][BN * BK];
    __shared__ __align__(16) unsigned short sWl[2][BN * BK];

    const int tid  = threadIdx.x;
    const int lane = tid & 31;
    const int wave = tid >> 5;
    const int wm   = wave & 3;    // 0..3 : 32-row M strip
    const int wn   = wave >> 2;   // 0..1 : 64-col N strip
    const int m0 = blockIdx.y * BM;
    const int n0 = blockIdx.x * BN;

    v8f acc[2][4];
    #pragma unroll
    for (int s = 0; s < 2; ++s)
        #pragma unroll
        for (int j = 0; j < 4; ++j)
            #pragma unroll
            for (int i = 0; i < 8; ++i) acc[s][j][i] = 0.0f;

    const int lsub  = lane & 15;
    const int khalf = (lane >> 4) * 8;

    // ---- prologue: stage chunk 0 into buffer 0 ----
    {
        float4 a[4], w[4];
        load_tile_f32(A, m0, 0, tid, a);
        if (PRECONV) issue_W_async(Whi, Wlo, sWh[0], sWl[0], n0, 0, tid);
        else         load_tile_f32(Wf, n0, 0, tid, w);
        store_split(a, sAh[0], sAl[0], tid);
        if (!PRECONV) store_split(w, sWh[0], sWl[0], tid);
        if (PRECONV) wait_async0();
        __syncthreads();
    }

    int kb = 0;
    for (int k0 = 0; k0 < K_DIM; k0 += BK, kb ^= 1) {
        const bool has_next = (k0 + BK) < K_DIM;
        float4 a[4], w[4];
        if (has_next) {
            // prefetch chunk k+1 into the other buffer (async DMA + reg loads)
            if (PRECONV) issue_W_async(Whi, Wlo, sWh[kb ^ 1], sWl[kb ^ 1], n0, k0 + BK, tid);
            else         load_tile_f32(Wf, n0, k0 + BK, tid, w);
            load_tile_f32(A, m0, k0 + BK, tid, a);
        }

        // ---- compute chunk k from buffer kb ----
        FragU ah[2], al[2];
        #pragma unroll
        for (int s = 0; s < 2; ++s) {
            const int arow = wm * 32 + s * 16 + lsub;
            ah[s].u[0] = *(const ush8*)&sAh[kb][arow * BK + khalf];
            ah[s].u[1] = *(const ush8*)&sAh[kb][arow * BK + 16 + khalf];
            al[s].u[0] = *(const ush8*)&sAl[kb][arow * BK + khalf];
            al[s].u[1] = *(const ush8*)&sAl[kb][arow * BK + 16 + khalf];
        }
        #pragma unroll
        for (int j = 0; j < 4; ++j) {
            FragU bh, bl;
            const int wrow = wn * 64 + j * 16 + lsub;
            bh.u[0] = *(const ush8*)&sWh[kb][wrow * BK + khalf];
            bh.u[1] = *(const ush8*)&sWh[kb][wrow * BK + 16 + khalf];
            bl.u[0] = *(const ush8*)&sWl[kb][wrow * BK + khalf];
            bl.u[1] = *(const ush8*)&sWl[kb][wrow * BK + 16 + khalf];
            #pragma unroll
            for (int s = 0; s < 2; ++s) {
                acc[s][j] = __builtin_amdgcn_wmma_f32_16x16x32_bf16(
                    false, ah[s].bf, false, bh.bf, (short)0, acc[s][j], false, false);
                acc[s][j] = __builtin_amdgcn_wmma_f32_16x16x32_bf16(
                    false, ah[s].bf, false, bl.bf, (short)0, acc[s][j], false, false);
                acc[s][j] = __builtin_amdgcn_wmma_f32_16x16x32_bf16(
                    false, al[s].bf, false, bh.bf, (short)0, acc[s][j], false, false);
            }
        }

        if (has_next) {
            // convert A (and W on fallback path) behind the WMMA stream
            store_split(a, sAh[kb ^ 1], sAl[kb ^ 1], tid);
            if (!PRECONV) store_split(w, sWh[kb ^ 1], sWl[kb ^ 1], tid);
            if (PRECONV) wait_async0();
        }
        __syncthreads();
    }

    // ---- epilogue: C layout VGPR r -> M=r (lanes 0-15) / r+8 (lanes 16-31) ----
    #pragma unroll
    for (int s = 0; s < 2; ++s) {
        const int mrow = m0 + wm * 32 + s * 16 + (lane >> 4) * 8;
        #pragma unroll
        for (int j = 0; j < 4; ++j) {
            const int n = n0 + wn * 64 + j * 16 + lsub;
            const float bv = bias[n];
            #pragma unroll
            for (int r = 0; r < 8; ++r) {
                C[(size_t)(mrow + r) * N_DIM + n] = acc[s][j][r] + bv;
            }
        }
    }
}

extern "C" void kernel_launch(void* const* d_in, const int* in_sizes, int n_in,
                              void* d_out, int out_size, void* d_ws, size_t ws_size,
                              hipStream_t stream) {
    (void)n_in; (void)out_size;
    const float* A = (const float*)d_in[0];   // data [32768,1024]
    const float* W = (const float*)d_in[1];   // W [1024,1024]
    const float* b = (const float*)d_in[2];   // bias [1024]
    float* C = (float*)d_out;
    const int Mtotal = in_sizes[0] / K_DIM;   // 32768
    dim3 grid(N_DIM / BN, Mtotal / BM);       // (8, 256)
    dim3 block(256);

    const size_t plane = (size_t)N_DIM * K_DIM;           // elements per plane
    if (ws_size >= plane * 2 * sizeof(unsigned short)) {
        unsigned short* Whi = (unsigned short*)d_ws;
        unsigned short* Wlo = Whi + plane;
        hipLaunchKernelGGL(wconvert, dim3((unsigned)(plane / 1024)), block, 0, stream,
                           W, Whi, Wlo);
        hipLaunchKernelGGL((groupfc_gemm<true>), grid, block, 0, stream,
                           A, W, Whi, Wlo, b, C);
    } else {
        hipLaunchKernelGGL((groupfc_gemm<false>), grid, block, 0, stream,
                           A, W, (const unsigned short*)nullptr,
                           (const unsigned short*)nullptr, b, C);
    }
}
